// PCGTConv_31817117728799
// MI455X (gfx1250) — matrix operand order repaired and checked by reading, not compiled, not run on hardware
//
#include <hip/hip_runtime.h>
#include <hip/hip_bf16.h>

#define NN     32768
#define INC    128
#define HID    256
#define NHEAD  4
#define NPART  64
#define SPART  512
#define NHID   1024     // H*HID
#define COLS3  3072     // 3*H*HID
#define SCALE_INV 0.0625f
#define EPS_LN 1e-5f
#define ALPHA_MIX 0.5f

typedef __bf16 bf16;
typedef __attribute__((ext_vector_type(16))) __bf16 v16bf;
typedef __attribute__((ext_vector_type(8)))  float  v8f;
typedef __attribute__((ext_vector_type(8)))  unsigned short us8;

union U16 { v16bf v; us8 h[2]; };
union F8  { v8f v; float f[8]; };

__device__ __forceinline__ unsigned short f2bfu(float x) {
  return __builtin_bit_cast(unsigned short, (__bf16)x);
}
__device__ __forceinline__ float bfu2f(unsigned short u) {
  return (float)__builtin_bit_cast(__bf16, u);
}
__device__ __forceinline__ us8 ldg8(const unsigned short* p) {
  return *reinterpret_cast<const us8*>(p);
}
// A operand (16x32 bf16): row-major row pointer at k-slice start.
// lanes 0-15: v0-7=k0..15 ; lanes 16-31: v0-7=k16..31
__device__ __forceinline__ v16bf loadA(const unsigned short* rowk, int g) {
  U16 u; u.h[0] = ldg8(rowk + 16*g); u.h[1] = ldg8(rowk + 16*g + 8); return u.v;
}
// B operand (32x16 bf16): per-lane 16 contraction-contiguous elements
// (caller passes pointer already offset by +16*g for the upper k half)
__device__ __forceinline__ v16bf loadB(const unsigned short* p) {
  U16 u; u.h[0] = ldg8(p); u.h[1] = ldg8(p + 8); return u.v;
}
__device__ __forceinline__ v8f wmma_bf(v16bf a, v16bf b, v8f c) {
  return __builtin_amdgcn_wmma_f32_16x16x32_bf16(false, a, false, b, (short)0, c, false, false);
}
__device__ __forceinline__ float redsum16(float x) {
  x += __shfl_xor(x, 1, 32); x += __shfl_xor(x, 2, 32);
  x += __shfl_xor(x, 4, 32); x += __shfl_xor(x, 8, 32); return x;
}
__device__ __forceinline__ float redmax16(float x) {
  x = fmaxf(x, __shfl_xor(x, 1, 32)); x = fmaxf(x, __shfl_xor(x, 2, 32));
  x = fmaxf(x, __shfl_xor(x, 4, 32)); x = fmaxf(x, __shfl_xor(x, 8, 32)); return x;
}
__device__ __forceinline__ float redsum32(float x) {
  x = redsum16(x); x += __shfl_xor(x, 16, 32); return x;
}

// ---------------- prep: transposed bf16 weight copies ----------------
__global__ void prep_weights(const float* __restrict__ wq, const float* __restrict__ wk,
                             const float* __restrict__ wv, const float* __restrict__ f0,
                             unsigned short* __restrict__ WT, unsigned short* __restrict__ F0T) {
  int o = blockIdx.x; int i = threadIdx.x;
  if (o < COLS3) {
    const float* src; int oc;
    if (o < NHID)        { src = wq; oc = o; }
    else if (o < 2*NHID) { src = wk; oc = o - NHID; }
    else                 { src = wv; oc = o - 2*NHID; }
    WT[(size_t)o*HID + i] = f2bfu(src[(size_t)i*NHID + oc]);
  } else {
    int o2 = o - COLS3;
    if (i < INC) F0T[o2*INC + i] = f2bfu(f0[(size_t)i*HID + o2]);
  }
}

__global__ void cvt_f2b(const float* __restrict__ src, unsigned short* __restrict__ dst, int n) {
  int i = blockIdx.x * blockDim.x + threadIdx.x;
  if (i < n) dst[i] = f2bfu(src[i]);
}

// ---------------- fc0 GEMM + LN0 + relu + pe ----------------
__global__ __launch_bounds__(32) void fc0_ln_kernel(
    const unsigned short* __restrict__ xin, const unsigned short* __restrict__ F0T,
    const float* __restrict__ fb, const float* __restrict__ g0, const float* __restrict__ b0,
    const float* __restrict__ pe, const int* __restrict__ labels,
    float* __restrict__ x0f, unsigned short* __restrict__ x0b) {
  int base = blockIdx.x * 16;
  int lane = threadIdx.x; int g = lane >> 4; int lr = lane & 15;
  v16bf a[4];
  const unsigned short* arow = xin + (size_t)(base + lr)*INC;
#pragma unroll
  for (int ks = 0; ks < 4; ++ks) a[ks] = loadA(arow + ks*32, g);

  v16bf bb[2][4];
#pragma unroll
  for (int ks = 0; ks < 4; ++ks)
    bb[0][ks] = loadB(F0T + (size_t)lr*INC + ks*32 + 16*g);

  F8 acc[16];
#pragma unroll
  for (int t = 0; t < 16; ++t) {
    if (t < 15) {
#pragma unroll
      for (int ks = 0; ks < 4; ++ks)
        bb[(t+1)&1][ks] = loadB(F0T + (size_t)((t+1)*16 + lr)*INC + ks*32 + 16*g);
    }
    v8f c = {};
#pragma unroll
    for (int ks = 0; ks < 4; ++ks) c = wmma_bf(a[ks], bb[t&1][ks], c);
    acc[t].v = c;
    float bias = fb[t*16 + lr];
#pragma unroll
    for (int v = 0; v < 8; ++v) acc[t].f[v] += bias;
  }
  float mean[8], rstd[8];
#pragma unroll
  for (int v = 0; v < 8; ++v) {
    float s = 0.f;
#pragma unroll
    for (int t = 0; t < 16; ++t) s += acc[t].f[v];
    mean[v] = redsum16(s) * (1.0f/HID);
  }
#pragma unroll
  for (int v = 0; v < 8; ++v) {
    float s = 0.f;
#pragma unroll
    for (int t = 0; t < 16; ++t) { float d = acc[t].f[v] - mean[v]; s += d*d; }
    rstd[v] = rsqrtf(redsum16(s)*(1.0f/HID) + EPS_LN);
  }
  int lab[8];
#pragma unroll
  for (int v = 0; v < 8; ++v) lab[v] = labels[base + v + 8*g];
#pragma unroll
  for (int t = 0; t < 16; ++t) {
    int col = t*16 + lr;
    float gg = g0[col], bb2 = b0[col];
#pragma unroll
    for (int v = 0; v < 8; ++v) {
      float x = (acc[t].f[v] - mean[v]) * rstd[v] * gg + bb2;
      x = fmaxf(x, 0.f);
      x += pe[lab[v]*HID + col];
      size_t idx = (size_t)(base + v + 8*g)*HID + col;
      x0f[idx] = x;
      x0b[idx] = f2bfu(x);
    }
  }
}

// ---------------- fused QKV GEMM + norm^2 / sum(K) reductions ----------------
__global__ __launch_bounds__(32) void qkv_kernel(
    const unsigned short* __restrict__ x0b, const unsigned short* __restrict__ WT,
    const float* __restrict__ qb, const float* __restrict__ kbias, const float* __restrict__ vb,
    unsigned short* __restrict__ Qbf, unsigned short* __restrict__ Kbf,
    unsigned short* __restrict__ Kt, unsigned short* __restrict__ Vt,
    float* __restrict__ Vf, float* __restrict__ ksum, float* __restrict__ stats) {
  int base = blockIdx.x * 16;
  int cb = blockIdx.y * 256;
  int lane = threadIdx.x; int g = lane >> 4; int lr = lane & 15;
  v16bf a[8];
  const unsigned short* arow = x0b + (size_t)(base + lr)*HID;
#pragma unroll
  for (int ks = 0; ks < 8; ++ks) a[ks] = loadA(arow + ks*32, g);

  v16bf bb[2][8];
#pragma unroll
  for (int ks = 0; ks < 8; ++ks)
    bb[0][ks] = loadB(WT + (size_t)(cb + lr)*HID + ks*32 + 16*g);

  int sect = blockIdx.y >> 2;   // 0=Q 1=K 2=V (uniform per block)
  float ssq = 0.f;
#pragma unroll
  for (int t = 0; t < 16; ++t) {
    if (t < 15) {
#pragma unroll
      for (int ks = 0; ks < 8; ++ks)
        bb[(t+1)&1][ks] = loadB(WT + (size_t)(cb + (t+1)*16 + lr)*HID + ks*32 + 16*g);
    }
    v8f c = {};
#pragma unroll
    for (int ks = 0; ks < 8; ++ks) c = wmma_bf(a[ks], bb[t&1][ks], c);
    F8 acc; acc.v = c;
    int cg = cb + t*16 + lr;
    int cl = cg - sect*NHID;
    float bias = (sect == 0) ? qb[cl] : (sect == 1) ? kbias[cl] : vb[cl];
#pragma unroll
    for (int v = 0; v < 8; ++v) acc.f[v] += bias;
    if (sect == 0) {
#pragma unroll
      for (int v = 0; v < 8; ++v) {
        int row = base + v + 8*g; float val = acc.f[v];
        Qbf[(size_t)row*NHID + cl] = f2bfu(val);
        ssq += val*val;
      }
    } else if (sect == 1) {
      float cs = 0.f;
#pragma unroll
      for (int v = 0; v < 8; ++v) {
        int row = base + v + 8*g; float val = acc.f[v];
        Kbf[(size_t)row*NHID + cl] = f2bfu(val);
        Kt[(size_t)cl*NN + row]    = f2bfu(val);
        ssq += val*val; cs += val;
      }
      atomicAdd(&ksum[cl], cs);
    } else {
#pragma unroll
      for (int v = 0; v < 8; ++v) {
        int row = base + v + 8*g; float val = acc.f[v];
        Vf[(size_t)row*NHID + cl] = val;
        Vt[(size_t)cl*NN + row]   = f2bfu(val);
      }
    }
  }
  if (sect < 2) {
    float tot = redsum32(ssq);
    if (lane == 0) atomicAdd(&stats[sect], tot);
  }
}

// ---------------- kvs^T = V^T K per head (split-K over nodes) ----------------
__global__ __launch_bounds__(32) void kvs_kernel(
    const unsigned short* __restrict__ Kt, const unsigned short* __restrict__ Vt,
    float* __restrict__ kvsTf) {
  int mt = blockIdx.x & 15, dt = blockIdx.x >> 4;
  int h = blockIdx.y;
  int nb0 = blockIdx.z * 1024;
  int lane = threadIdx.x; int g = lane >> 4; int lr = lane & 15;
  const unsigned short* Arow = Vt + (size_t)(h*HID + dt*16 + lr)*NN + nb0;
  const unsigned short* Brow = Kt + (size_t)(h*HID + mt*16 + lr)*NN + nb0;
  v8f c = {};
  // depth-2 software pipeline over the 32 k-steps
  v16bf aC = loadA(Arow, g),        bC = loadB(Brow + 16*g);
  v16bf aN = loadA(Arow + 32, g),   bN = loadB(Brow + 32 + 16*g);
#pragma unroll 4
  for (int kk = 0; kk < 32; ++kk) {
    v16bf aF = aC, bF = bC;
    if (kk < 30) {
      aF = loadA(Arow + (kk+2)*32, g);
      bF = loadB(Brow + (kk+2)*32 + 16*g);
    }
    c = wmma_bf(aC, bC, c);
    aC = aN; bC = bN; aN = aF; bN = bF;
  }
  F8 acc; acc.v = c;
#pragma unroll
  for (int v = 0; v < 8; ++v)
    atomicAdd(&kvsTf[(size_t)(h*HID + dt*16 + v + 8*g)*HID + mt*16 + lr], acc.f[v]);
}

// ---------------- partition flash attention (per p,h, 16-row tile) ----------------
__global__ __launch_bounds__(32) void part_attn_kernel(
    const int* __restrict__ pidx,
    const unsigned short* __restrict__ Qbf, const unsigned short* __restrict__ Kbf,
    const unsigned short* __restrict__ Vt, float* __restrict__ xpart) {
  __shared__ unsigned short plds[512];   // 16x32 bf16 prob tile
  int mt = blockIdx.x; int h = blockIdx.y; int p = blockIdx.z;
  int lane = threadIdx.x; int g = lane >> 4; int lr = lane & 15;
  const int* pp = pidx + p*SPART;
  int rowQ = pp[mt*16 + lr];
  const unsigned short* Qrow = Qbf + (size_t)rowQ*NHID + h*HID;
  v16bf aq[8];
#pragma unroll
  for (int ks = 0; ks < 8; ++ks) aq[ks] = loadA(Qrow + ks*32, g);
  v8f zero = {};
  F8 O[16];
#pragma unroll
  for (int t = 0; t < 16; ++t) O[t].v = zero;
  float mrun[8], lrun[8];
#pragma unroll
  for (int v = 0; v < 8; ++v) { mrun[v] = -3.0e38f; lrun[v] = 0.f; }
  int key0 = pp[0];  // nodes in a partition are consecutive (arange layout)
  for (int kb = 0; kb < 16; ++kb) {
    int k0 = pp[kb*32 + lr];
    int k1 = pp[kb*32 + 16 + lr];
    const unsigned short* Kr0 = Kbf + (size_t)k0*NHID + h*HID;
    const unsigned short* Kr1 = Kbf + (size_t)k1*NHID + h*HID;
    v8f s0v = {}, s1v = {};
    v16bf b0c = loadB(Kr0 + 16*g);
    v16bf b1c = loadB(Kr1 + 16*g);
#pragma unroll
    for (int ks = 0; ks < 8; ++ks) {
      v16bf b0n = b0c, b1n = b1c;
      if (ks < 7) {
        b0n = loadB(Kr0 + (ks+1)*32 + 16*g);
        b1n = loadB(Kr1 + (ks+1)*32 + 16*g);
      }
      s0v = wmma_bf(aq[ks], b0c, s0v);
      s1v = wmma_bf(aq[ks], b1c, s1v);
      b0c = b0n; b1c = b1n;
    }
    F8 s0, s1; s0.v = s0v; s1.v = s1v;
    float alpha[8];
#pragma unroll
    for (int v = 0; v < 8; ++v) {
      float a0 = s0.f[v]*SCALE_INV, a1 = s1.f[v]*SCALE_INV;
      s0.f[v] = a0; s1.f[v] = a1;
      float rm = redmax16(fmaxf(a0, a1));
      float mnew = fmaxf(mrun[v], rm);
      alpha[v] = __expf(mrun[v] - mnew);
      mrun[v] = mnew;
    }
#pragma unroll
    for (int v = 0; v < 8; ++v) {
      float p0 = __expf(s0.f[v] - mrun[v]);
      float p1 = __expf(s1.f[v] - mrun[v]);
      s0.f[v] = p0; s1.f[v] = p1;
      lrun[v] = lrun[v]*alpha[v] + redsum16(p0 + p1);
    }
#pragma unroll
    for (int t = 0; t < 16; ++t) {
#pragma unroll
      for (int v = 0; v < 8; ++v) O[t].f[v] *= alpha[v];
    }
    __syncthreads();
#pragma unroll
    for (int v = 0; v < 8; ++v) {
      plds[(v + 8*g)*32 + lr]      = f2bfu(s0.f[v]);
      plds[(v + 8*g)*32 + 16 + lr] = f2bfu(s1.f[v]);
    }
    __syncthreads();
    U16 ap;
    ap.h[0] = *reinterpret_cast<const us8*>(&plds[lr*32 + 16*g]);
    ap.h[1] = *reinterpret_cast<const us8*>(&plds[lr*32 + 16*g + 8]);
    v16bf aP = ap.v;
    int kvt = key0 + kb*32 + 16*g;
    v16bf bvc = loadB(Vt + (size_t)(h*HID + lr)*NN + kvt);
#pragma unroll
    for (int t = 0; t < 16; ++t) {
      v16bf bvn = bvc;
      if (t < 15)
        bvn = loadB(Vt + (size_t)(h*HID + (t+1)*16 + lr)*NN + kvt);
      O[t].v = wmma_bf(aP, bvc, O[t].v);
      bvc = bvn;
    }
  }
  float invl[8];
#pragma unroll
  for (int v = 0; v < 8; ++v) invl[v] = 0.25f / lrun[v];   // head-mean folded in
  int outn[8];
#pragma unroll
  for (int v = 0; v < 8; ++v) outn[v] = pp[mt*16 + v + 8*g];
#pragma unroll
  for (int t = 0; t < 16; ++t) {
    int col = t*16 + lr;
#pragma unroll
    for (int v = 0; v < 8; ++v)
      atomicAdd(&xpart[(size_t)outn[v]*HID + col], O[t].f[v]*invl[v]);
  }
}

// ---------------- global num GEMM + combine + residual + LN1 ----------------
__global__ __launch_bounds__(32) void final_kernel(
    const unsigned short* __restrict__ Qbf, const unsigned short* __restrict__ kvsb,
    const float* __restrict__ ksum, const float* __restrict__ stats,
    const float* __restrict__ Vf, const float* __restrict__ xpart,
    const float* __restrict__ x0f, const float* __restrict__ gamma_logit,
    const float* __restrict__ g1, const float* __restrict__ b1,
    float* __restrict__ out) {
  int base = blockIdx.x * 16;
  int lane = threadIdx.x; int g = lane >> 4; int lr = lane & 15;
  float s = rsqrtf(stats[0]) * rsqrtf(stats[1]);  // 1/(||Q||*||K||)
  float gamma = 1.f / (1.f + __expf(-gamma_logit[0]));
  float wg = (1.f - gamma) * 0.25f;               // head-mean folded in
  F8 acc[16];
#pragma unroll
  for (int t = 0; t < 16; ++t) {
    int col = t*16 + lr;
#pragma unroll
    for (int v = 0; v < 8; ++v)
      acc[t].f[v] = gamma * xpart[(size_t)(base + v + 8*g)*HID + col];
  }
  for (int h = 0; h < NHEAD; ++h) {
    const unsigned short* Qrow = Qbf + (size_t)(base + lr)*NHID + h*HID;
    v16bf aq[8];
#pragma unroll
    for (int ks = 0; ks < 8; ++ks) aq[ks] = loadA(Qrow + ks*32, g);
    float dot = 0.f;
    const float* kscol = ksum + h*HID;
    for (int d = 0; d < HID; ++d) dot += bfu2f(Qrow[d]) * kscol[d];
    float invden[8];
#pragma unroll
    for (int v = 0; v < 8; ++v) {
      float dd = __shfl(dot, v + 8*g, 32);
      invden[v] = 1.f / (s*dd + (float)NN);
    }
#pragma unroll
    for (int t = 0; t < 16; ++t) {
      const unsigned short* brow = kvsb + (size_t)(h*HID + t*16 + lr)*HID;
      v8f c = {};
      v16bf bcur = loadB(brow + 16*g);
#pragma unroll
      for (int ks = 0; ks < 8; ++ks) {
        v16bf bn = bcur;
        if (ks < 7) bn = loadB(brow + (ks+1)*32 + 16*g);
        c = wmma_bf(aq[ks], bcur, c);
        bcur = bn;
      }
      F8 num; num.v = c;
      int col = t*16 + lr;
#pragma unroll
      for (int v = 0; v < 8; ++v) {
        float vv = Vf[(size_t)(base + v + 8*g)*NHID + h*HID + col];
        acc[t].f[v] += wg * (s*num.f[v] + (float)NN * vv) * invden[v];
      }
    }
  }
#pragma unroll
  for (int t = 0; t < 16; ++t) {
    int col = t*16 + lr;
#pragma unroll
    for (int v = 0; v < 8; ++v) {
      float x0v = x0f[(size_t)(base + v + 8*g)*HID + col];
      acc[t].f[v] = ALPHA_MIX*acc[t].f[v] + (1.f - ALPHA_MIX)*x0v;
    }
  }
  float mean[8], rstd[8];
#pragma unroll
  for (int v = 0; v < 8; ++v) {
    float ssum = 0.f;
#pragma unroll
    for (int t = 0; t < 16; ++t) ssum += acc[t].f[v];
    mean[v] = redsum16(ssum) * (1.0f/HID);
  }
#pragma unroll
  for (int v = 0; v < 8; ++v) {
    float ssum = 0.f;
#pragma unroll
    for (int t = 0; t < 16; ++t) { float d = acc[t].f[v]-mean[v]; ssum += d*d; }
    rstd[v] = rsqrtf(redsum16(ssum)*(1.0f/HID) + EPS_LN);
  }
#pragma unroll
  for (int t = 0; t < 16; ++t) {
    int col = t*16 + lr;
    float gg = g1[col], bb = b1[col];
#pragma unroll
    for (int v = 0; v < 8; ++v)
      out[(size_t)(base + v + 8*g)*HID + col] = (acc[t].f[v]-mean[v])*rstd[v]*gg + bb;
  }
}

extern "C" void kernel_launch(void* const* d_in, const int* in_sizes, int n_in,
                              void* d_out, int out_size, void* d_ws, size_t ws_size,
                              hipStream_t stream) {
  (void)in_sizes; (void)n_in; (void)out_size; (void)ws_size;
  const float* node_feat = (const float*)d_in[0];
  const int*   pidx      = (const int*)  d_in[1];
  const int*   labels    = (const int*)  d_in[2];
  const float* fc0_w = (const float*)d_in[3];
  const float* fc0_b = (const float*)d_in[4];
  const float* ln0_g = (const float*)d_in[5];
  const float* ln0_b = (const float*)d_in[6];
  const float* pe    = (const float*)d_in[7];
  const float* wq_w = (const float*)d_in[8];
  const float* wq_b = (const float*)d_in[9];
  const float* wk_w = (const float*)d_in[10];
  const float* wk_b = (const float*)d_in[11];
  const float* wv_w = (const float*)d_in[12];
  const float* wv_b = (const float*)d_in[13];
  const float* gamma_logit = (const float*)d_in[14];
  const float* ln1_g = (const float*)d_in[15];
  const float* ln1_b = (const float*)d_in[16];
  float* out = (float*)d_out;

  char* w = (char*)d_ws;
  auto alloc = [&](size_t bytes) { char* p = w; w += (bytes + 255) & ~(size_t)255; return p; };
  unsigned short* xin  = (unsigned short*)alloc((size_t)NN*INC*2);
  unsigned short* F0T  = (unsigned short*)alloc((size_t)HID*INC*2);
  unsigned short* WT   = (unsigned short*)alloc((size_t)COLS3*HID*2);
  float*          x0f  = (float*)         alloc((size_t)NN*HID*4);
  unsigned short* x0b  = (unsigned short*)alloc((size_t)NN*HID*2);
  unsigned short* Qbf  = (unsigned short*)alloc((size_t)NN*NHID*2);
  unsigned short* Kbf  = (unsigned short*)alloc((size_t)NN*NHID*2);
  unsigned short* Kt   = (unsigned short*)alloc((size_t)NHID*NN*2);
  unsigned short* Vt   = (unsigned short*)alloc((size_t)NHID*NN*2);
  float*          Vf   = (float*)         alloc((size_t)NN*NHID*4);
  float*          kvsf = (float*)         alloc((size_t)NHEAD*HID*HID*4);
  unsigned short* kvsb = (unsigned short*)alloc((size_t)NHEAD*HID*HID*2);
  float*          ksum = (float*)         alloc((size_t)NHID*4);
  float*          stat = (float*)         alloc(256);
  float*          xprt = (float*)         alloc((size_t)NN*HID*4);

  hipMemsetAsync(kvsf, 0, (size_t)NHEAD*HID*HID*4, stream);
  hipMemsetAsync(ksum, 0, (size_t)NHID*4, stream);
  hipMemsetAsync(stat, 0, 256, stream);
  hipMemsetAsync(xprt, 0, (size_t)NN*HID*4, stream);

  prep_weights<<<dim3(COLS3 + HID), 256, 0, stream>>>(wq_w, wk_w, wv_w, fc0_w, WT, F0T);
  cvt_f2b<<<dim3((NN*INC)/256), 256, 0, stream>>>(node_feat, xin, NN*INC);
  fc0_ln_kernel<<<dim3(NN/16), 32, 0, stream>>>(xin, F0T, fc0_b, ln0_g, ln0_b, pe, labels, x0f, x0b);
  qkv_kernel<<<dim3(NN/16, 12), 32, 0, stream>>>(x0b, WT, wq_b, wk_b, wv_b, Qbf, Kbf, Kt, Vt, Vf, ksum, stat);
  kvs_kernel<<<dim3(256, NHEAD, 32), 32, 0, stream>>>(Kt, Vt, kvsf);
  cvt_f2b<<<dim3((NHEAD*HID*HID)/256), 256, 0, stream>>>(kvsf, kvsb, NHEAD*HID*HID);
  part_attn_kernel<<<dim3(SPART/16, NHEAD, NPART), 32, 0, stream>>>(pidx, Qbf, Kbf, Vt, xprt);
  final_kernel<<<dim3(NN/16), 32, 0, stream>>>(Qbf, kvsb, ksum, stat, Vf, xprt, x0f, gamma_logit, ln1_g, ln1_b, out);
}
